// TritonCastleAttention_44092134260894
// MI455X (gfx1250) — matrix operand order repaired and checked by reading, not compiled, not run on hardware
//
#include <hip/hip_runtime.h>
#include <hip/hip_bf16.h>
#include <stdint.h>

typedef __bf16 bf16;
typedef __attribute__((ext_vector_type(16))) __bf16 v16bf;
typedef __attribute__((ext_vector_type(8)))  __bf16 v8bf;
typedef __attribute__((ext_vector_type(8)))  float  v8f;

#define SEQ    2048
#define DIM    1024
#define HEADS  16
#define DHEAD  64
#define QKV_N  (6 * DIM)

#define BM 128
#define BN 128
#define BK 32
#define LSTR 40          // 32 + 8 bf16 pad: keeps 16B alignment, breaks bank conflicts
#define TILE_ELEMS (BM * LSTR)
#define NTHREADS 256     // 8 waves (wave32)

__device__ inline v16bf cat16(v8bf lo, v8bf hi) {
  return __builtin_shufflevector(lo, hi, 0,1,2,3,4,5,6,7,8,9,10,11,12,13,14,15);
}

// CDNA5 async copy: 16B global -> LDS, no VGPR staging, tracked by ASYNCcnt.
__device__ inline void async_ld16(unsigned lds_addr, const void* gaddr) {
  asm volatile("global_load_async_to_lds_b128 %0, %1, off"
               :: "v"(lds_addr), "v"(gaddr) : "memory");
}

// NT GEMM core: acc(128x128 block) += A[row_base.., k0:k1] * B[col_base.., k0:k1]^T
// A: bf16 row-major [*, lda]; B: bf16 row-major [Nrows, ldb] (B^T's columns as rows).
// B rows >= Nrows are address-clamped; their products land only in discarded columns.
// Double-buffered LDS fed by global_load_async_to_lds_b128 (4 issues/wave/K-step).
// 8 waves: wave (wm 0..3, wn 0..1) owns 32x64 = 2x4 WMMA tiles.
__device__ void gemm_core(const bf16* __restrict__ A, int lda,
                          const bf16* __restrict__ B, int ldb, int Nrows,
                          int k0, int k1, int row_base, int col_base,
                          bf16* ls /* [2 bufs][A,B][TILE_ELEMS] */, v8f acc[2][4])
{
  const int t     = threadIdx.x;
  const int lane  = t & 31;
  const int wave  = t >> 5;
  const int wm    = wave >> 1;
  const int wn    = wave & 1;
  const int lrow  = lane & 15;
  const int khalf = lane >> 4;
  const int ldr   = t >> 2;          // 0..63: LDS fill row
  const int ldc8  = (t & 3) << 3;    // 0,8,16,24: 8 bf16 (16B) per thread

  auto issue = [&](int buf, int kk) {
#pragma unroll
    for (int p = 0; p < 2; ++p) {
      const int r = p * 64 + ldr;
      bf16* dA = ls + (size_t)(buf * 2 + 0) * TILE_ELEMS + r * LSTR + ldc8;
      async_ld16((unsigned)(size_t)dA,
                 A + (size_t)(row_base + r) * lda + kk + ldc8);
      int gn = col_base + r;
      if (gn >= Nrows) gn = Nrows - 1;   // clamp: keeps ASYNCcnt accounting uniform
      bf16* dB = ls + (size_t)(buf * 2 + 1) * TILE_ELEMS + r * LSTR + ldc8;
      async_ld16((unsigned)(size_t)dB,
                 B + (size_t)gn * ldb + kk + ldc8);
    }
  };

  issue(0, k0);
  int buf = 0;
  for (int kk = k0; kk < k1; kk += BK) {
    if (kk + BK < k1) {
      issue(buf ^ 1, kk + BK);
      asm volatile("s_wait_asynccnt 4" ::: "memory");  // prev tile (in-order) landed
    } else {
      asm volatile("s_wait_asynccnt 0" ::: "memory");
    }
    __syncthreads();   // all waves' copies of current tile are in LDS

    const bf16* lsA = ls + (size_t)(buf * 2 + 0) * TILE_ELEMS;
    const bf16* lsB = ls + (size_t)(buf * 2 + 1) * TILE_ELEMS;

    // A fragment: lane<16 -> M=lrow, K {0..7,16..23}; lane>=16 -> K {8..15,24..31}
    v16bf afrag[2];
#pragma unroll
    for (int mt = 0; mt < 2; ++mt) {
      const bf16* pa = lsA + (wm * 32 + mt * 16 + lrow) * LSTR + khalf * 8;
      afrag[mt] = cat16(*(const v8bf*)pa, *(const v8bf*)(pa + 16));
    }
    // B fragment: lane<16 -> N=lrow, K {0..7,16..23}; lane>=16 -> K {8..15,24..31}
    v16bf bfrag[4];
#pragma unroll
    for (int nt = 0; nt < 4; ++nt) {
      const bf16* pb = lsB + (wn * 64 + nt * 16 + lrow) * LSTR + khalf * 8;
      bfrag[nt] = cat16(*(const v8bf*)pb, *(const v8bf*)(pb + 16));
    }
#pragma unroll
    for (int mt = 0; mt < 2; ++mt)
#pragma unroll
      for (int nt = 0; nt < 4; ++nt)
        acc[mt][nt] = __builtin_amdgcn_wmma_f32_16x16x32_bf16(
            false, afrag[mt], false, bfrag[nt], (short)0, acc[mt][nt],
            false, false);

    __syncthreads();   // tile consumed by all waves before it is overwritten
    buf ^= 1;
  }
}

#define GEMM_PROLOGUE(AQ, LDA, BQ, LDB, NR, K0, K1)                         \
  __shared__ bf16 lsmem[2 * 2 * TILE_ELEMS];                                \
  const int row_base = blockIdx.y * BM;                                     \
  const int col_base = blockIdx.x * BN;                                     \
  v8f acc[2][4];                                                            \
  {                                                                         \
    v8f z = {0.f, 0.f, 0.f, 0.f, 0.f, 0.f, 0.f, 0.f};                       \
    for (int i = 0; i < 2; ++i)                                             \
      for (int j = 0; j < 4; ++j) acc[i][j] = z;                            \
  }                                                                         \
  gemm_core(AQ, LDA, BQ, LDB, NR, K0, K1, row_base, col_base, lsmem, acc);

// C tile element: row = row_base + wm*32 + mt*16 + (lane>>4)*8 + r ; col = .. + (lane&15)
#define EPILOGUE(...)                                                       \
  {                                                                         \
    const int lane = threadIdx.x & 31;                                      \
    const int wave = threadIdx.x >> 5;                                      \
    const int wm = wave >> 1, wn = wave & 1;                                \
    _Pragma("unroll") for (int mt = 0; mt < 2; ++mt)                        \
    _Pragma("unroll") for (int nt = 0; nt < 4; ++nt)                        \
    _Pragma("unroll") for (int r = 0; r < 8; ++r) {                         \
      const int row = row_base + wm * 32 + mt * 16 + (lane >> 4) * 8 + r;   \
      const int col = col_base + wn * 64 + nt * 16 + (lane & 15);           \
      float v = acc[mt][nt][r];                                             \
      __VA_ARGS__;                                                          \
    }                                                                       \
  }

// ---------------- prep kernels ----------------
__global__ void k_cvt(const float* __restrict__ in, bf16* __restrict__ out, int n) {
  int i = blockIdx.x * blockDim.x + threadIdx.x;
  if (i < n) out[i] = (bf16)in[i];
}

__global__ void k_tr_cvt(const float* __restrict__ in, bf16* __restrict__ out,
                         int rows, int cols) {
  int i = blockIdx.x * blockDim.x + threadIdx.x;
  if (i < rows * cols) {
    int r = i / cols, c = i % cols;
    out[(size_t)c * rows + r] = (bf16)in[i];
  }
}

// ---------------- GEMM kernels ----------------
// qkv = x @ W_qkv ; scatter into head-major qu/ku/vu/qc/kc/vc (scale on qu,qc), plus vc^T
__global__ __launch_bounds__(NTHREADS)
void k_qkv(const bf16* __restrict__ xb, const bf16* __restrict__ Wt,
           bf16* __restrict__ qkv, bf16* __restrict__ vcT) {
  GEMM_PROLOGUE(xb, DIM, Wt, DIM, QKV_N, 0, DIM);
  EPILOGUE(
    int which = col >> 10;           // 0=qu 1=ku 2=vu 3=qc 4=kc 5=vc
    int head  = (col >> 6) & 15;
    int d     = col & 63;
    float s = (which == 0 || which == 3) ? v * 0.125f : v;  // DIM_HEAD^-0.5
    qkv[(((size_t)which * HEADS + head) * SEQ + row) * DHEAD + d] = (bf16)s;
    if (which == 5)
      vcT[((size_t)head * DHEAD + d) * SEQ + row] = (bf16)v
  );
}

// term1 = tril(qc_s @ vu^T)
__global__ __launch_bounds__(NTHREADS)
void k_term1(const bf16* __restrict__ qc, const bf16* __restrict__ vu,
             bf16* __restrict__ T) {
  GEMM_PROLOGUE(qc, DHEAD, vu, DHEAD, SEQ, 0, DHEAD);
  EPILOGUE(T[(size_t)row * SEQ + col] = (col <= row) ? (bf16)v : (bf16)0.0f);
}

// lookahead = strict_upper(sigmoid(qu_s @ ku^T))
__global__ __launch_bounds__(NTHREADS)
void k_look(const bf16* __restrict__ qu, const bf16* __restrict__ ku,
            bf16* __restrict__ L) {
  GEMM_PROLOGUE(qu, DHEAD, ku, DHEAD, SEQ, 0, DHEAD);
  EPILOGUE(
    float g = (col > row) ? (1.0f / (1.0f + __expf(-v))) : 0.0f;
    L[(size_t)row * SEQ + col] = (bf16)g
  );
}

// scores = qc_s @ kc^T, -1e30 above diagonal
__global__ __launch_bounds__(NTHREADS)
void k_sc(const bf16* __restrict__ qc, const bf16* __restrict__ kc,
          float* __restrict__ S) {
  GEMM_PROLOGUE(qc, DHEAD, kc, DHEAD, SEQ, 0, DHEAD);
  EPILOGUE(S[(size_t)row * SEQ + col] = (col > row) ? -1.0e30f : v);
}

// scores -= silu(T @ L^T); skip blocks above the diagonal, clamp K-loop to the
// nonzero band [bj*128, (bi+1)*128) implied by T lower- / L upper-triangularity.
__global__ __launch_bounds__(NTHREADS)
void k_su(const bf16* __restrict__ T, const bf16* __restrict__ L,
          float* __restrict__ S) {
  const int bi = blockIdx.y, bj = blockIdx.x;
  if (bj > bi) return;  // already -1e30 from k_sc
  GEMM_PROLOGUE(T, SEQ, L, SEQ, SEQ, bj * BN, (bi + 1) * BM);
  EPILOGUE(
    if (col <= row) {
      size_t idx = (size_t)row * SEQ + col;
      float sc = S[idx];
      S[idx] = sc - v / (1.0f + __expf(-v));   // sc - silu(v)
    }
  );
}

// row softmax, P written as bf16 (into the L buffer, reused)
__global__ __launch_bounds__(NTHREADS)
void k_softmax(const float* __restrict__ S, bf16* __restrict__ P) {
  const int row = blockIdx.x;
  const float* s = S + (size_t)row * SEQ;
  __shared__ float red[NTHREADS];
  const int t = threadIdx.x;

  float m = -1.0e30f;
  for (int c = t; c < SEQ; c += NTHREADS) m = fmaxf(m, s[c]);
  red[t] = m; __syncthreads();
  for (int o = NTHREADS / 2; o > 0; o >>= 1) {
    if (t < o) red[t] = fmaxf(red[t], red[t + o]);
    __syncthreads();
  }
  m = red[0]; __syncthreads();

  float sum = 0.f;
  for (int c = t; c < SEQ; c += NTHREADS) sum += __expf(s[c] - m);
  red[t] = sum; __syncthreads();
  for (int o = NTHREADS / 2; o > 0; o >>= 1) {
    if (t < o) red[t] += red[t + o];
    __syncthreads();
  }
  const float inv = 1.0f / red[0];
  for (int c = t; c < SEQ; c += NTHREADS)
    P[(size_t)row * SEQ + c] = (bf16)(__expf(s[c] - m) * inv);
}

// out_h = P @ vc  (as P @ (vc^T)^T, N = 64; clamped B rows feed discarded cols)
__global__ __launch_bounds__(NTHREADS)
void k_pv(const bf16* __restrict__ P, const bf16* __restrict__ vcT,
          bf16* __restrict__ O, int col_off) {
  GEMM_PROLOGUE(P, SEQ, vcT, SEQ, DHEAD, 0, SEQ);
  EPILOGUE(
    if (col < DHEAD)
      O[(size_t)row * DIM + col_off + col] = (bf16)v
  );
}

// final: out = attn_out @ W_out   (f32 result)
__global__ __launch_bounds__(NTHREADS)
void k_out(const bf16* __restrict__ Oi, const bf16* __restrict__ Wot,
           float* __restrict__ out) {
  GEMM_PROLOGUE(Oi, DIM, Wot, DIM, DIM, 0, DIM);
  EPILOGUE(out[(size_t)row * DIM + col] = v);
}

// ---------------- launch ----------------
extern "C" void kernel_launch(void* const* d_in, const int* in_sizes, int n_in,
                              void* d_out, int out_size, void* d_ws, size_t ws_size,
                              hipStream_t stream) {
  const float* x    = (const float*)d_in[0];
  const float* Wqkv = (const float*)d_in[1];
  const float* Wout = (const float*)d_in[2];
  float* out = (float*)d_out;

  char* ws = (char*)d_ws;
  size_t off = 0;
  auto take = [&](size_t bytes) -> char* {
    char* p = ws + off;
    off = (off + bytes + 255) & ~(size_t)255;
    return p;
  };
  bf16*  xb  = (bf16*)take((size_t)SEQ * DIM * 2);
  bf16*  Wt  = (bf16*)take((size_t)QKV_N * DIM * 2);       // W_qkv^T
  bf16*  Wot = (bf16*)take((size_t)DIM * DIM * 2);         // W_out^T
  bf16*  qkv = (bf16*)take((size_t)6 * HEADS * SEQ * DHEAD * 2);
  bf16*  vcT = (bf16*)take((size_t)HEADS * DHEAD * SEQ * 2);
  bf16*  Tb  = (bf16*)take((size_t)SEQ * SEQ * 2);
  bf16*  Lb  = (bf16*)take((size_t)SEQ * SEQ * 2);         // also reused for P
  float* Sb  = (float*)take((size_t)SEQ * SEQ * 4);
  bf16*  Ob  = (bf16*)take((size_t)SEQ * DIM * 2);

  k_cvt<<<(SEQ * DIM + 255) / 256, 256, 0, stream>>>(x, xb, SEQ * DIM);
  k_tr_cvt<<<(DIM * QKV_N + 255) / 256, 256, 0, stream>>>(Wqkv, Wt, DIM, QKV_N);
  k_tr_cvt<<<(DIM * DIM + 255) / 256, 256, 0, stream>>>(Wout, Wot, DIM, DIM);

  k_qkv<<<dim3(QKV_N / BN, SEQ / BM), NTHREADS, 0, stream>>>(xb, Wt, qkv, vcT);

  const dim3 gAttn(SEQ / BN, SEQ / BM);
  for (int h = 0; h < HEADS; ++h) {
    const bf16* qu = qkv + (((size_t)0 * HEADS + h) * SEQ) * DHEAD;
    const bf16* ku = qkv + (((size_t)1 * HEADS + h) * SEQ) * DHEAD;
    const bf16* vu = qkv + (((size_t)2 * HEADS + h) * SEQ) * DHEAD;
    const bf16* qc = qkv + (((size_t)3 * HEADS + h) * SEQ) * DHEAD;
    const bf16* kc = qkv + (((size_t)4 * HEADS + h) * SEQ) * DHEAD;

    k_term1<<<gAttn, NTHREADS, 0, stream>>>(qc, vu, Tb);
    k_look <<<gAttn, NTHREADS, 0, stream>>>(qu, ku, Lb);
    k_sc   <<<gAttn, NTHREADS, 0, stream>>>(qc, kc, Sb);
    k_su   <<<gAttn, NTHREADS, 0, stream>>>(Tb, Lb, Sb);
    k_softmax<<<SEQ, NTHREADS, 0, stream>>>(Sb, Lb);
    k_pv<<<dim3(1, SEQ / BM), NTHREADS, 0, stream>>>(
        Lb, vcT + (size_t)h * DHEAD * SEQ, Ob, h * DHEAD);
  }

  k_out<<<dim3(DIM / BN, SEQ / BM), NTHREADS, 0, stream>>>(Ob, Wot, out);
}
